// KSpaceCTGNN_19215683682946
// MI455X (gfx1250) — matrix operand order repaired and checked by reading, not compiled, not run on hardware
//
#include <hip/hip_runtime.h>

// ---------------- problem constants (match reference) ----------------
constexpr int N_NODES = 10000;
constexpr int N_EDGES = 80000;
constexpr int N_GRAPH = 16;
constexpr int D_IN    = 64;
constexpr int D_HID   = 128;
constexpr int HEADS   = 8;
constexpr int LAYERS  = 4;
constexpr int HC      = HEADS * D_HID;     // 1024
constexpr int QKV_W   = 3 * HC;            // 3072
constexpr int CAT_W   = QKV_W + D_HID;     // 3200

typedef __bf16 bf16_t;
typedef __attribute__((ext_vector_type(16))) __bf16 v16bf;
typedef __attribute__((ext_vector_type(8)))  __bf16 v8bf;
typedef __attribute__((ext_vector_type(4)))  __bf16 v4bf;
typedef __attribute__((ext_vector_type(8)))  float  v8f;

// ---------------- gfx1250 async global->LDS copy ----------------
typedef int v4i __attribute__((vector_size(4 * sizeof(int))));
typedef __attribute__((address_space(1))) v4i* as1_v4i_ptr;
typedef __attribute__((address_space(3))) v4i* as3_v4i_ptr;

__device__ __forceinline__ void asyncLoadB128(void* lds_dst, const bf16_t* gsrc) {
#if __has_builtin(__builtin_amdgcn_global_load_async_to_lds_b128)
    __builtin_amdgcn_global_load_async_to_lds_b128(
        (as1_v4i_ptr)(uintptr_t)gsrc,
        (as3_v4i_ptr)(unsigned int)(uintptr_t)lds_dst,
        0, 0);
#else
    unsigned int loff = (unsigned int)(uintptr_t)lds_dst;
    asm volatile("global_load_async_to_lds_b128 %0, %1, off"
                 :: "v"(loff), "v"(gsrc) : "memory");
#endif
}

__device__ __forceinline__ void waitAsync0() {
#if __has_builtin(__builtin_amdgcn_s_wait_asynccnt)
    __builtin_amdgcn_s_wait_asynccnt(0);
#else
    asm volatile("s_wait_asynccnt 0" ::: "memory");
#endif
}

// ---------------- helpers ----------------
__device__ __forceinline__ void atomicMaxFloat(float* addr, float value) {
    if (value >= 0.0f) atomicMax((int*)addr, __float_as_int(value));
    else               atomicMin((unsigned int*)addr, __float_as_uint(value));
}

__device__ __forceinline__ float waveReduceAdd(float v) {
    #pragma unroll
    for (int m = 16; m >= 1; m >>= 1) v += __shfl_xor(v, m, 32);
    return v;
}

// ---------------- conversion / packing ----------------
__global__ void cvt_f32_to_bf16_kernel(const float* __restrict__ src,
                                       bf16_t* __restrict__ dst, int n) {
    int i = blockIdx.x * blockDim.x + threadIdx.x;
    if (i < n) dst[i] = (bf16_t)src[i];
}

// Wp [64 x 128] -> WpT [128 x 64] bf16
__global__ void pack_wp_kernel(const float* __restrict__ Wp, bf16_t* __restrict__ WpT) {
    int i = blockIdx.x * blockDim.x + threadIdx.x;
    if (i >= D_HID * D_IN) return;
    int o = i / D_IN, k = i % D_IN;
    WpT[i] = (bf16_t)Wp[k * D_HID + o];
}

// Wq/Wk/Wv [L][128][1024], Wskip [L][128][128] -> WT [L][3200][128] bf16 (transposed)
// biases -> biasCat [L][3200] fp32
__global__ void pack_weights_kernel(const float* __restrict__ Wq, const float* __restrict__ Wk,
                                    const float* __restrict__ Wv, const float* __restrict__ Ws,
                                    const float* __restrict__ bq, const float* __restrict__ bk,
                                    const float* __restrict__ bv, const float* __restrict__ bs,
                                    bf16_t* __restrict__ WT, float* __restrict__ biasCat) {
    long long i = (long long)blockIdx.x * blockDim.x + threadIdx.x;
    long long total = (long long)LAYERS * CAT_W * D_HID;
    if (i >= total) return;
    int k   = (int)(i % D_HID);
    long long r = i / D_HID;
    int col = (int)(r % CAT_W);
    int l   = (int)(r / CAT_W);
    float w;
    if      (col < HC)     w = Wq[((long long)l * D_HID + k) * HC + col];
    else if (col < 2 * HC) w = Wk[((long long)l * D_HID + k) * HC + (col - HC)];
    else if (col < QKV_W)  w = Wv[((long long)l * D_HID + k) * HC + (col - 2 * HC)];
    else                   w = Ws[((long long)l * D_HID + k) * D_HID + (col - QKV_W)];
    WT[i] = (bf16_t)w;
    if (k == 0) {
        float b;
        if      (col < HC)     b = bq[l * HC + col];
        else if (col < 2 * HC) b = bk[l * HC + col - HC];
        else if (col < QKV_W)  b = bv[l * HC + col - 2 * HC];
        else                   b = bs[l * D_HID + col - QKV_W];
        biasCat[l * CAT_W + col] = b;
    }
}

// ---------------- WMMA GEMM with async-LDS staged B ----------------
// C[MxNc] = A[MxK] * BT[NcxK]^T + bias.  A row-major bf16, BT row-major bf16
// (each BT row = one output column, K long).  Block = 8 waves; block tile =
// 128(M) x 64(N).  The 64xK B tile is async-copied to LDS once and reused by
// all 8 waves; each wave computes 16x64 via 4 fp32 accumulators.  Per k-step
// all 4 B fragments are hoisted into registers before the WMMA cluster so the
// 4 matrix ops issue back-to-back behind a single dscnt wait.
template <bool OUT_BF16, int K>
__global__ __launch_bounds__(256)
void wmma_gemm_lds_kernel(const bf16_t* __restrict__ A, const bf16_t* __restrict__ BT,
                          const float* __restrict__ bias,
                          float* __restrict__ Cf, bf16_t* __restrict__ Cb,
                          int mTiles, int nTiles, int ldc) {
    constexpr int LSTR = K + 8;                      // +16B/row: kill LDS bank conflicts
    __shared__ __align__(16) bf16_t Bsm[64 * LSTR];

    const int nTile = blockIdx.x % nTiles;
    const int mBlk  = blockIdx.x / nTiles;
    const int nBase = nTile * 64;

    // cooperative async copy of B tile [64 x K] into LDS (16B chunks per lane)
    constexpr int KC = K / 8;                        // 16B chunks per row
    constexpr int CH = 64 * KC;                      // total chunks
    for (int c = threadIdx.x; c < CH; c += 256) {
        int col = c / KC, kp = c % KC;
        asyncLoadB128(&Bsm[col * LSTR + kp * 8],
                      BT + (size_t)(nBase + col) * K + kp * 8);
    }
    waitAsync0();
    __syncthreads();

    const int lane  = threadIdx.x & 31;
    const int wib   = threadIdx.x >> 5;
    const int mTile = mBlk * 8 + wib;
    if (mTile >= mTiles) return;                     // wave-uniform: EXEC stays full

    const int lhalf = lane >> 4;                     // 0 or 1
    const int l15   = lane & 15;

    int col[4];
    v8f acc[4];
    #pragma unroll
    for (int j = 0; j < 4; ++j) {
        col[j] = nBase + j * 16 + l15;
        float b = bias[col[j]];
        #pragma unroll
        for (int i = 0; i < 8; ++i) acc[j][i] = b;
    }

    // A fragment per ISA: lanes 0-15 carry K = {0..7,16..23}, lanes 16-31 K = {8..15,24..31}
    const bf16_t* aRow = A + (size_t)(mTile * 16 + l15) * K + lhalf * 8;
    // B fragment base in LDS: lanes 0-15 carry K = {0..7,16..23}, lanes 16-31 K = {8..15,24..31}
    const bf16_t* bBase = &Bsm[l15 * LSTR + lhalf * 8];

    #pragma unroll
    for (int k = 0; k < K; k += 32) {
        v8bf alo = *(const v8bf*)(aRow + k);
        v8bf ahi = *(const v8bf*)(aRow + k + 16);
        v16bf av;
        #pragma unroll
        for (int i = 0; i < 8; ++i) { av[i] = alo[i]; av[i + 8] = ahi[i]; }

        v16bf bf[4];
        #pragma unroll
        for (int j = 0; j < 4; ++j) {
            const bf16_t* bRow = bBase + j * 16 * LSTR + k;
            v8bf blo = *(const v8bf*)(bRow);
            v8bf bhi = *(const v8bf*)(bRow + 16);
            #pragma unroll
            for (int i = 0; i < 8; ++i) { bf[j][i] = blo[i]; bf[j][i + 8] = bhi[i]; }
        }
        #pragma unroll
        for (int j = 0; j < 4; ++j) {
            acc[j] = __builtin_amdgcn_wmma_f32_16x16x32_bf16(
                false, av, false, bf[j], (short)0, acc[j], false, false);
        }
    }

    // C/D layout: lane<16 -> M = base+i, lane>=16 -> M = base+8+i; N = col[j]
    const int mOff = mTile * 16 + lhalf * 8;
    #pragma unroll
    for (int j = 0; j < 4; ++j) {
        #pragma unroll
        for (int i = 0; i < 8; ++i) {
            if (OUT_BF16) Cb[(size_t)(mOff + i) * ldc + col[j]] = (bf16_t)acc[j][i];
            else          Cf[(size_t)(mOff + i) * ldc + col[j]] = acc[j][i];
        }
    }
}

// ---------------- attention buffers init ----------------
__global__ void attn_init_kernel(float* __restrict__ mbuf, float* __restrict__ denom,
                                 float* __restrict__ outAvg, int nH, int nD) {
    int i = blockIdx.x * blockDim.x + threadIdx.x;
    if (i < nH) { mbuf[i] = -3.402823466e+38f; denom[i] = 0.0f; }
    if (i < nD) outAvg[i] = 0.0f;
}

// ---------------- edge scores + segment max (wave per (edge, head)) ----------------
__global__ __launch_bounds__(256)
void edge_score_max_kernel(const bf16_t* __restrict__ qkv, const int* __restrict__ ei,
                           float* __restrict__ scores, float* __restrict__ mbuf,
                           float scale) {
    const int lane = threadIdx.x & 31;
    const int wave = blockIdx.x * (blockDim.x >> 5) + (threadIdx.x >> 5);
    if (wave >= N_EDGES * HEADS) return;
    const int e = wave >> 3;
    const int h = wave & 7;
    const int src = ei[e];
    const int dst = ei[N_EDGES + e];
    const bf16_t* qp = qkv + (size_t)dst * QKV_W + h * D_HID;        // query @ target
    const bf16_t* kp = qkv + (size_t)src * QKV_W + HC + h * D_HID;   // key   @ source
    v4bf q4 = *(const v4bf*)(qp + lane * 4);
    v4bf k4 = *(const v4bf*)(kp + lane * 4);
    float s = 0.0f;
    #pragma unroll
    for (int i = 0; i < 4; ++i) s += (float)q4[i] * (float)k4[i];
    s = waveReduceAdd(s) * scale;
    if (lane == 0) {
        scores[(size_t)e * HEADS + h] = s;
        atomicMaxFloat(&mbuf[(size_t)dst * HEADS + h], s);
    }
}

// ---------------- exp(score - max) + segment sum ----------------
__global__ void edge_exp_sum_kernel(const int* __restrict__ ei, float* __restrict__ scores,
                                    const float* __restrict__ mbuf, float* __restrict__ denom) {
    int i = blockIdx.x * blockDim.x + threadIdx.x;
    if (i >= N_EDGES * HEADS) return;
    int e = i >> 3, h = i & 7;
    int dst = ei[N_EDGES + e];
    float ev = __expf(scores[i] - mbuf[(size_t)dst * HEADS + h]);
    scores[i] = ev;
    atomicAdd(&denom[(size_t)dst * HEADS + h], ev);
}

// ---------------- message scatter: out[dst] += mean_h(alpha * v[src,h,:]) ----------------
__global__ __launch_bounds__(256)
void edge_msg_kernel(const bf16_t* __restrict__ qkv, const int* __restrict__ ei,
                     const float* __restrict__ scores, const float* __restrict__ denom,
                     float* __restrict__ outAvg) {
    const int e = blockIdx.x;
    const int t = threadIdx.x;
    const int src = ei[e];
    const int dst = ei[N_EDGES + e];
    const int c  = t & 127;
    const int h0 = (t >> 7) * 4;                 // thread covers 4 heads -> 1 atomic
    const bf16_t* vp = qkv + (size_t)src * QKV_W + 2 * HC;
    float sum = 0.0f;
    #pragma unroll
    for (int hh = 0; hh < 4; ++hh) {
        int h = h0 + hh;
        float alpha = scores[(size_t)e * HEADS + h] / denom[(size_t)dst * HEADS + h];
        sum += alpha * (float)vp[h * D_HID + c];
    }
    atomicAdd(&outAvg[(size_t)dst * D_HID + c], sum * 0.125f);   // /HEADS folded in
}

// ---------------- beta-gate + LayerNorm + ReLU (wave per node), writes fp32 + bf16 h ----------------
__global__ __launch_bounds__(256)
void gate_ln_kernel(const float* __restrict__ outAvg, const float* __restrict__ xs,
                    const float* __restrict__ Wb, const float* __restrict__ lng,
                    const float* __restrict__ lnb, float* __restrict__ h,
                    bf16_t* __restrict__ hb) {
    const int lane = threadIdx.x & 31;
    const int n = blockIdx.x * (blockDim.x >> 5) + (threadIdx.x >> 5);
    if (n >= N_NODES) return;
    float o[4], x4[4];
    float dot = 0.0f;
    #pragma unroll
    for (int i = 0; i < 4; ++i) {
        int c = lane + i * 32;
        float ov = outAvg[(size_t)n * D_HID + c];
        float xv = xs[(size_t)n * D_HID + c];
        o[i] = ov; x4[i] = xv;
        dot += ov * Wb[c] + xv * Wb[D_HID + c] + (ov - xv) * Wb[2 * D_HID + c];
    }
    dot = waveReduceAdd(dot);
    float gate = 1.0f / (1.0f + __expf(-dot));
    float val[4], mean = 0.0f;
    #pragma unroll
    for (int i = 0; i < 4; ++i) { val[i] = gate * x4[i] + (1.0f - gate) * o[i]; mean += val[i]; }
    mean = waveReduceAdd(mean) * (1.0f / D_HID);
    float var = 0.0f;
    #pragma unroll
    for (int i = 0; i < 4; ++i) { float d = val[i] - mean; var += d * d; }
    var = waveReduceAdd(var) * (1.0f / D_HID);
    float rstd = rsqrtf(var + 1e-5f);
    #pragma unroll
    for (int i = 0; i < 4; ++i) {
        int c = lane + i * 32;
        float y = (val[i] - mean) * rstd * lng[c] + lnb[c];
        y = fmaxf(y, 0.0f);
        h[(size_t)n * D_HID + c]  = y;
        hb[(size_t)n * D_HID + c] = (bf16_t)y;
    }
}

// ---------------- global mean pool ----------------
__global__ void pool_zero_kernel(float* __restrict__ sums, float* __restrict__ cnt) {
    int i = blockIdx.x * blockDim.x + threadIdx.x;
    if (i < N_GRAPH * D_HID) sums[i] = 0.0f;
    if (i < N_GRAPH) cnt[i] = 0.0f;
}
__global__ void pool_acc_kernel(const float* __restrict__ h, const int* __restrict__ batch,
                                float* __restrict__ sums, float* __restrict__ cnt) {
    int i = blockIdx.x * blockDim.x + threadIdx.x;
    if (i >= N_NODES * D_HID) return;
    int n = i >> 7, c = i & 127;
    int b = batch[n];
    atomicAdd(&sums[b * D_HID + c], h[i]);
    if (c == 0) atomicAdd(&cnt[b], 1.0f);
}
__global__ void pool_div_kernel(const float* __restrict__ sums, const float* __restrict__ cnt,
                                float* __restrict__ out) {
    int i = blockIdx.x * blockDim.x + threadIdx.x;
    if (i >= N_GRAPH * D_HID) return;
    out[i] = sums[i] / fmaxf(cnt[i >> 7], 1.0f);
}

// ---------------- launcher ----------------
extern "C" void kernel_launch(void* const* d_in, const int* in_sizes, int n_in,
                              void* d_out, int out_size, void* d_ws, size_t ws_size,
                              hipStream_t stream) {
    const float* x      = (const float*)d_in[0];
    const int*   ei     = (const int*)d_in[1];
    const int*   batch  = (const int*)d_in[2];
    const float* Wp     = (const float*)d_in[3];
    const float* bp     = (const float*)d_in[4];
    const float* Wq     = (const float*)d_in[5];
    const float* bq     = (const float*)d_in[6];
    const float* Wk     = (const float*)d_in[7];
    const float* bk     = (const float*)d_in[8];
    const float* Wv     = (const float*)d_in[9];
    const float* bv     = (const float*)d_in[10];
    const float* Wskip  = (const float*)d_in[11];
    const float* bskip  = (const float*)d_in[12];
    const float* Wbeta  = (const float*)d_in[13];
    const float* ln_g   = (const float*)d_in[14];
    const float* ln_b   = (const float*)d_in[15];
    float* out = (float*)d_out;

    // workspace carve-up (256B aligned)
    char* ws = (char*)d_ws;
    size_t off = 0;
    auto carve = [&](size_t bytes) -> void* {
        void* p = ws + off;
        off = (off + bytes + 255) & ~(size_t)255;
        return p;
    };
    bf16_t* xb      = (bf16_t*)carve((size_t)N_NODES * D_IN * 2);
    bf16_t* WpT     = (bf16_t*)carve((size_t)D_HID * D_IN * 2);
    bf16_t* WT      = (bf16_t*)carve((size_t)LAYERS * CAT_W * D_HID * 2);
    float*  biasCat = (float*)carve((size_t)LAYERS * CAT_W * 4);
    float*  h       = (float*)carve((size_t)N_NODES * D_HID * 4);
    bf16_t* hb      = (bf16_t*)carve((size_t)N_NODES * D_HID * 2);
    bf16_t* qkv     = (bf16_t*)carve((size_t)N_NODES * QKV_W * 2);
    float*  xs      = (float*)carve((size_t)N_NODES * D_HID * 4);
    float*  scores  = (float*)carve((size_t)N_EDGES * HEADS * 4);
    float*  mbuf    = (float*)carve((size_t)N_NODES * HEADS * 4);
    float*  denom   = (float*)carve((size_t)N_NODES * HEADS * 4);
    float*  outAvg  = (float*)carve((size_t)N_NODES * D_HID * 4);
    float*  sums    = (float*)carve((size_t)N_GRAPH * D_HID * 4);
    float*  cnt     = (float*)carve((size_t)N_GRAPH * 4);

    const int TB = 256;
    auto blocks = [](long long n, int tb) { return (int)((n + tb - 1) / tb); };

    const int mTiles = N_NODES / 16;          // 625
    const int mBlks  = (mTiles + 7) / 8;      // 79

    // 1) convert x and pack weights to bf16 (transposed for WMMA B)
    cvt_f32_to_bf16_kernel<<<blocks((long long)N_NODES * D_IN, TB), TB, 0, stream>>>(
        x, xb, N_NODES * D_IN);
    pack_wp_kernel<<<blocks(D_HID * D_IN, TB), TB, 0, stream>>>(Wp, WpT);
    pack_weights_kernel<<<blocks((long long)LAYERS * CAT_W * D_HID, TB), TB, 0, stream>>>(
        Wq, Wk, Wv, Wskip, bq, bk, bv, bskip, WT, biasCat);

    // 2) initial projection: h = x @ Wp + bp   (M=10000, K=64, Nc=128)
    wmma_gemm_lds_kernel<false, D_IN><<<mBlks * (D_HID / 64), TB, 0, stream>>>(
        xb, WpT, bp, h, nullptr, mTiles, D_HID / 64, D_HID);
    cvt_f32_to_bf16_kernel<<<blocks((long long)N_NODES * D_HID, TB), TB, 0, stream>>>(
        h, hb, N_NODES * D_HID);

    const float scale = 0.088388347648318447f;   // 1/sqrt(128)

    for (int l = 0; l < LAYERS; ++l) {
        const bf16_t* WTl = WT + (size_t)l * CAT_W * D_HID;
        const float*  bCl = biasCat + (size_t)l * CAT_W;

        // qkv = h @ [Wq|Wk|Wv] + b  -> bf16 [N x 3072]
        wmma_gemm_lds_kernel<true, D_HID><<<mBlks * (QKV_W / 64), TB, 0, stream>>>(
            hb, WTl, bCl, nullptr, qkv, mTiles, QKV_W / 64, QKV_W);
        // xs = h @ Wskip + bskip -> fp32 [N x 128]
        wmma_gemm_lds_kernel<false, D_HID><<<mBlks * (D_HID / 64), TB, 0, stream>>>(
            hb, WTl + (size_t)QKV_W * D_HID, bCl + QKV_W, xs, nullptr,
            mTiles, D_HID / 64, D_HID);

        attn_init_kernel<<<blocks((long long)N_NODES * D_HID, TB), TB, 0, stream>>>(
            mbuf, denom, outAvg, N_NODES * HEADS, N_NODES * D_HID);

        edge_score_max_kernel<<<N_EDGES, TB, 0, stream>>>(qkv, ei, scores, mbuf, scale);
        edge_exp_sum_kernel<<<blocks((long long)N_EDGES * HEADS, TB), TB, 0, stream>>>(
            ei, scores, mbuf, denom);
        edge_msg_kernel<<<N_EDGES, TB, 0, stream>>>(qkv, ei, scores, denom, outAvg);

        gate_ln_kernel<<<blocks((long long)N_NODES * 32, TB), TB, 0, stream>>>(
            outAvg, xs, Wbeta + (size_t)l * 3 * D_HID,
            ln_g + (size_t)l * D_HID, ln_b + (size_t)l * D_HID, h, hb);
    }

    // 3) global mean pool
    pool_zero_kernel<<<blocks(N_GRAPH * D_HID, TB), TB, 0, stream>>>(sums, cnt);
    pool_acc_kernel<<<blocks((long long)N_NODES * D_HID, TB), TB, 0, stream>>>(
        h, batch, sums, cnt);
    pool_div_kernel<<<blocks(N_GRAPH * D_HID, TB), TB, 0, stream>>>(sums, cnt, out);
}
